// GATv2_DX_DG_DR_H4_77747497992606
// MI455X (gfx1250) — matrix operand order repaired and checked by reading, not compiled, not run on hardware
//
#include <hip/hip_runtime.h>
#include <hip/hip_bf16.h>

// ---------------- problem constants (from reference) ----------------
#define NN   25600
#define EE   102400
#define BB   512
#define HH   4
#define CC   256
#define DD   1024      // H*C
#define XDD  64
#define EDD  16
#define GDD  200
#define RDD  100

typedef __attribute__((ext_vector_type(16))) __bf16 bf16x16;
typedef __attribute__((ext_vector_type(8)))  __bf16 bf16x8;
typedef __attribute__((ext_vector_type(8)))  float  f32x8;

typedef __attribute__((ext_vector_type(4))) unsigned tdmv4u;
typedef __attribute__((ext_vector_type(8))) int      tdmv8i;
typedef __attribute__((ext_vector_type(4))) int      tdmv4i;

#define HAVE_TDM __has_builtin(__builtin_amdgcn_tensor_load_to_lds)

// ---------------- helpers ----------------
__device__ __forceinline__ __bf16 f2bf(float f) {
    unsigned u = __float_as_uint(f);
    u += 0x7FFFu + ((u >> 16) & 1u);          // round-to-nearest-even
    unsigned short hs = (unsigned short)(u >> 16);
    __bf16 r;
    __builtin_memcpy(&r, &hs, sizeof(r));
    return r;
}

// order-preserving float<->uint encoding so segment-max can use atomicMax(u32)
__device__ __forceinline__ unsigned encodeF(float x) {
    unsigned u = __float_as_uint(x);
    return (u & 0x80000000u) ? ~u : (u | 0x80000000u);
}
__device__ __forceinline__ float decodeF(unsigned v) {
    unsigned u = (v & 0x80000000u) ? (v & 0x7FFFFFFFu) : ~v;
    return __uint_as_float(u);
}
__device__ __forceinline__ bool finitef(float x) {
    return (__float_as_uint(x) & 0x7F800000u) != 0x7F800000u;
}

#define SHUF16(lo, hi) __builtin_shufflevector(lo, hi, 0,1,2,3,4,5,6,7,8,9,10,11,12,13,14,15)

// ---------------- zero init ----------------
__global__ void k_zero(unsigned* __restrict__ p, long n) {
    long i = (long)blockIdx.x * blockDim.x + threadIdx.x;
    if (i < n) p[i] = 0u;
}

// ---------------- one-time bf16 conversions ----------------
__global__ void k_cvt_bf16(const float* __restrict__ in, __bf16* __restrict__ out, long n) {
    long i = (long)blockIdx.x * blockDim.x + threadIdx.x;
    if (i < n) out[i] = f2bf(in[i]);
}
// Wt[d][k] = bf16(W[k][d])  (weights transposed so B-operand K runs contiguous)
__global__ void k_cvt_transpose_bf16(const float* __restrict__ W, __bf16* __restrict__ Wt,
                                     int K, int Do) {
    long i = (long)blockIdx.x * blockDim.x + threadIdx.x;
    if (i >= (long)K * Do) return;
    const int k = (int)(i / Do), d = (int)(i % Do);
    Wt[(long)d * K + k] = f2bf(W[i]);
}

// ---------------- bf16 WMMA GEMM: Y[M,Do] = Xb[M,K] @ Wt[Do,K]^T + bias ----------------
// grid = (Do/128, M/16); block = 128 (4 waves). Each wave: one 16x32 output tile.
// The 128x32 bf16 B panel per K-step is staged in LDS via the Tensor Data Mover
// (double-buffered, s_wait_tensorcnt pipelined), falling back to cooperative copy.
__global__ void __launch_bounds__(128)
gemm_bias_wmma(const __bf16* __restrict__ Xb, const __bf16* __restrict__ Wt,
               const float* __restrict__ bias, float* __restrict__ Y,
               int M, int K, int Do) {
    __shared__ __bf16 sB[2][128 * 32];                 // 16 KB double buffer

    const int t    = threadIdx.x;
    const int wIn  = t >> 5;
    const int lane = t & 31;
    const int half = lane >> 4;                        // ISA lane-half
    const int l16  = lane & 15;

    const int tm    = blockIdx.y << 4;                 // 16 output rows / block
    const int tnBlk = blockIdx.x << 7;                 // 128 output cols / block
    const __bf16* __restrict__ arow = Xb + (long)(tm + l16) * K;

    f32x8 c0 = {}, c1 = {};

#if HAVE_TDM
    const unsigned ldsBase = (unsigned)(size_t)(const void*)&sB[0][0];
    auto stage = [&](int bsel, int k0) {               // TDM: 2D tile 32(k) x 128(cols)
        if (t < 32) {
            const unsigned long long ga =
                (unsigned long long)(size_t)(const void*)(Wt + (long)tnBlk * K + k0);
            tdmv4u g0; tdmv8i g1; tdmv4i gz; tdmv8i gz8;
            g0[0] = 1u;                                             // count=1
            g0[1] = ldsBase + (unsigned)bsel * (128u * 32u * 2u);   // lds_addr
            g0[2] = (unsigned)(ga & 0xFFFFFFFFull);                 // global_addr lo
            g0[3] = (unsigned)((ga >> 32) & 0x01FFFFFFull) | (2u << 30); // addr hi | type=2
            g1[0] = 0x00010000;                                     // data_size = 2B
            g1[1] = (int)((unsigned)(K & 0xFFFF) << 16);            // tensor_dim0 lo
            g1[2] = (int)(((unsigned)(Do & 0xFFFF) << 16) | ((unsigned)K >> 16));
            g1[3] = (int)((32u << 16) | ((unsigned)Do >> 16));      // tile_dim0 = 32
            g1[4] = 128;                                            // tile_dim1 = 128 rows
            g1[5] = K;                                              // tensor_dim0_stride
            g1[6] = 0; g1[7] = 0;
            gz[0] = 0; gz[1] = 0; gz[2] = 0; gz[3] = 0;
            gz8[0] = 0; gz8[1] = 0; gz8[2] = 0; gz8[3] = 0;
            gz8[4] = 0; gz8[5] = 0; gz8[6] = 0; gz8[7] = 0;
            // 6-arg toolchain form: (g0 v4u, g1 v8i, v4i, v4i, v8i, cpol)
            __builtin_amdgcn_tensor_load_to_lds(g0, g1, gz, gz, gz8, 0);
        }
    };
#else
    auto stage = [&](int bsel, int k0) {               // fallback: 1 row (=output col) / thread
        const __bf16* srcp = Wt + (long)(tnBlk + t) * K + k0;
        __bf16* dstp = &sB[bsel][t * 32];
        *(bf16x8*)(dstp + 0)  = *(const bf16x8*)(srcp + 0);
        *(bf16x8*)(dstp + 8)  = *(const bf16x8*)(srcp + 8);
        *(bf16x8*)(dstp + 16) = *(const bf16x8*)(srcp + 16);
        *(bf16x8*)(dstp + 24) = *(const bf16x8*)(srcp + 24);
    };
#endif

    int bsel = 0;
    stage(0, 0);
    for (int k0 = 0; k0 < K; k0 += 32) {
#if HAVE_TDM && __has_builtin(__builtin_amdgcn_s_wait_tensorcnt)
        if (t < 32) __builtin_amdgcn_s_wait_tensorcnt(0);
#endif
        __syncthreads();
        if (k0 + 32 < K) stage(bsel ^ 1, k0 + 32);     // prefetch next panel

        // ---- A operand 16x32 bf16 (global, L2-resident; contiguous b128 loads)
        // lanes 0-15: elems0-7 = K k0..+7,  elems8-15 = K k0+16..+23 ; lanes16-31 shifted by 8
        __builtin_prefetch(arow + k0 + 64, 0, 0);
        const bf16x8 alo = *(const bf16x8*)(arow + k0 + half * 8);
        const bf16x8 ahi = *(const bf16x8*)(arow + k0 + 16 + half * 8);
        const bf16x16 a = SHUF16(alo, ahi);

        // ---- B operands 32x16 bf16 (LDS): lane col = r, K same order as A (k0..+7 / k0+16..+23, shifted by 8 for lanes16-31)
        const __bf16* sb = &sB[bsel][0];
        const int r0 = (wIn << 5) + l16;
        const int r1 = r0 + 16;
        const bf16x8 b0lo = *(const bf16x8*)(sb + r0 * 32 + half * 8);
        const bf16x8 b0hi = *(const bf16x8*)(sb + r0 * 32 + 16 + half * 8);
        const bf16x8 b1lo = *(const bf16x8*)(sb + r1 * 32 + half * 8);
        const bf16x8 b1hi = *(const bf16x8*)(sb + r1 * 32 + 16 + half * 8);
        const bf16x16 b0 = SHUF16(b0lo, b0hi);
        const bf16x16 b1 = SHUF16(b1lo, b1hi);

        c0 = __builtin_amdgcn_wmma_f32_16x16x32_bf16(false, a, false, b0, (short)0, c0, false, false);
        c1 = __builtin_amdgcn_wmma_f32_16x16x32_bf16(false, a, false, b1, (short)0, c1, false, false);

        __syncthreads();
        bsel ^= 1;
    }

    // C layout: col = tn(+16)+l16 ; row = tm + half*8 + r
    const int bcol0 = tnBlk + (wIn << 5) + l16;
    const int bcol1 = bcol0 + 16;
    const float bv0 = bias[bcol0];
    const float bv1 = bias[bcol1];
#pragma unroll
    for (int r = 0; r < 8; ++r) {
        const long row = tm + half * 8 + r;
        Y[row * (long)Do + bcol0] = c0[r] + bv0;
        Y[row * (long)Do + bcol1] = c1[r] + bv1;
    }
}

// ---------------- per-edge logits + segment max ----------------
__global__ void k_edge_logits(const int* __restrict__ src,
                              const int* __restrict__ dst,
                              const float* __restrict__ ea,     // [E,16]
                              const float* __restrict__ We,     // [16,1024]
                              const float* __restrict__ att,    // [1024]
                              const float* __restrict__ xl,     // [N,1024]
                              const float* __restrict__ xr,     // [N,1024]
                              float* __restrict__ logits,       // [E,4]
                              unsigned* __restrict__ segmax) {  // [N,4] encoded
    const int e = blockIdx.x;
    const int t = threadIdx.x;
    __shared__ float sea[EDD];
    __shared__ float partial[8];
    if (t < EDD) sea[t] = ea[(long)e * EDD + t];
    __syncthreads();

    const int s  = src[e];
    const int d0 = dst[e];
    const int h  = t >> 6;

    float sum = 0.f;
#pragma unroll
    for (int j = 0; j < 4; ++j) {
        const int d = t * 4 + j;
        float eev = 0.f;
#pragma unroll
        for (int k = 0; k < EDD; ++k) eev += sea[k] * We[k * DD + d];
        float m = xl[(long)s * DD + d] + xr[(long)d0 * DD + d] + eev;
        m = m > 0.f ? m : 0.2f * m;                   // leaky_relu(0.2)
        sum += m * att[d];
    }
#pragma unroll
    for (int off = 16; off; off >>= 1) sum += __shfl_xor(sum, off, 32);
    if ((t & 31) == 0) partial[t >> 5] = sum;
    __syncthreads();
    if ((t & 63) == 0) {
        const float tot = partial[t >> 5] + partial[(t >> 5) + 1];
        logits[e * HH + h] = tot;
        atomicMax(&segmax[d0 * HH + h], encodeF(tot));
    }
}

// ---------------- exp(logit - max) + segment sum ----------------
__global__ void k_edge_exp(const int* __restrict__ dst,
                           const float* __restrict__ logits,
                           const unsigned* __restrict__ segmax,
                           float* __restrict__ expv,
                           float* __restrict__ segsum) {
    const int i = blockIdx.x * blockDim.x + threadIdx.x;
    if (i >= EE * HH) return;
    const int e = i >> 2, h = i & 3;
    const int d0 = dst[e];
    float m = decodeF(segmax[d0 * HH + h]);
    if (!finitef(m)) m = 0.f;                         // empty-segment semantics
    const float ev = __expf(logits[i] - m);
    expv[i] = ev;
    atomicAdd(&segsum[d0 * HH + h], ev);
}

// ---------------- weighted scatter: out[dst] += alpha * xl[src] ----------------
__global__ void k_edge_aggregate(const int* __restrict__ src,
                                 const int* __restrict__ dst,
                                 const float* __restrict__ expv,
                                 const float* __restrict__ segsum,
                                 const float* __restrict__ xl,
                                 float* __restrict__ out) {
    const int e = blockIdx.x;
    const int t = threadIdx.x;
    const int s  = src[e];
    const int d0 = dst[e];
    const int h  = t >> 6;
    const float alpha = expv[e * HH + h] / (segsum[d0 * HH + h] + 1e-16f);
#pragma unroll
    for (int j = 0; j < 4; ++j) {
        const int d = t * 4 + j;
        atomicAdd(&out[(long)d0 * DD + d], alpha * xl[(long)s * DD + d]);
    }
}

// ---------------- h = elu(acc + bias) ----------------
__global__ void k_node_finish(float* __restrict__ h, const float* __restrict__ bias, long n) {
    long i = (long)blockIdx.x * blockDim.x + threadIdx.x;
    if (i >= n) return;
    float v = h[i] + bias[i & (DD - 1)];
    h[i] = v > 0.f ? v : (__expf(v) - 1.f);
}

// ---------------- gate logits: gl[n] = h[n]·Wg + bg ; batch max ----------------
__global__ void k_gate(const float* __restrict__ h, const float* __restrict__ Wg,
                       const float* __restrict__ bg, const int* __restrict__ batch,
                       float* __restrict__ gl, unsigned* __restrict__ bmax) {
    const int n = blockIdx.x * (blockDim.x >> 5) + (threadIdx.x >> 5);
    if (n >= NN) return;
    const int lane = threadIdx.x & 31;
    float sum = 0.f;
#pragma unroll
    for (int k = 0; k < DD / 32; ++k) {
        const int d = lane + 32 * k;
        sum += h[(long)n * DD + d] * Wg[d];
    }
#pragma unroll
    for (int off = 16; off; off >>= 1) sum += __shfl_xor(sum, off, 32);
    if (lane == 0) {
        const float v = sum + bg[0];
        gl[n] = v;
        atomicMax(&bmax[batch[n]], encodeF(v));
    }
}

__global__ void k_gate_exp(const int* __restrict__ batch, const float* __restrict__ gl,
                           const unsigned* __restrict__ bmax,
                           float* __restrict__ ge, float* __restrict__ bsum) {
    const int i = blockIdx.x * blockDim.x + threadIdx.x;
    if (i >= NN) return;
    const int b = batch[i];
    float m = decodeF(bmax[b]);
    if (!finitef(m)) m = 0.f;
    const float ev = __expf(gl[i] - m);
    ge[i] = ev;
    atomicAdd(&bsum[b], ev);
}

// ---------------- pooled[b] += gate * h[n] ----------------
__global__ void k_pool(const int* __restrict__ batch, const float* __restrict__ ge,
                       const float* __restrict__ bsum, const float* __restrict__ h,
                       float* __restrict__ pooled) {
    const int n = blockIdx.x;
    const int t = threadIdx.x;
    const int b = batch[n];
    const float gate = ge[n] / (bsum[b] + 1e-16f);
#pragma unroll
    for (int j = 0; j < 4; ++j) {
        const int d = t * 4 + j;
        atomicAdd(&pooled[(long)b * DD + d], gate * h[(long)n * DD + d]);
    }
}

// ---------------- final fc over concat(pooled, g, r) ----------------
__global__ void k_final(const float* __restrict__ pooled, const float* __restrict__ g,
                        const float* __restrict__ r, const float* __restrict__ Wfc,
                        const float* __restrict__ bfc, float* __restrict__ out) {
    const int b = blockIdx.x;
    const int t = threadIdx.x;
    const int ZD = DD + GDD + RDD;                    // 1324
    float local = 0.f;
    for (int i = t; i < ZD; i += 256) {
        float z;
        if (i < DD)            z = pooled[(long)b * DD + i];
        else if (i < DD + GDD) z = g[(long)b * GDD + (i - DD)];
        else                   z = r[(long)b * RDD + (i - DD - GDD)];
        local += z * Wfc[i];
    }
    __shared__ float red[256];
    red[t] = local;
    __syncthreads();
    for (int s2 = 128; s2; s2 >>= 1) {
        if (t < s2) red[t] += red[t + s2];
        __syncthreads();
    }
    if (t == 0) out[b] = red[0] + bfc[0];
}

// ==================== launcher ====================
extern "C" void kernel_launch(void* const* d_in, const int* in_sizes, int n_in,
                              void* d_out, int out_size, void* d_ws, size_t ws_size,
                              hipStream_t stream) {
    const float* x    = (const float*)d_in[0];
    const int*   ei   = (const int*)  d_in[1];
    const int*   src  = ei;
    const int*   dst  = ei + EE;
    const float* ea   = (const float*)d_in[2];
    const int*   bat  = (const int*)  d_in[3];
    const float* g    = (const float*)d_in[4];
    const float* r    = (const float*)d_in[5];
    const float* Wl1  = (const float*)d_in[6];
    const float* bl1  = (const float*)d_in[7];
    const float* Wr1  = (const float*)d_in[8];
    const float* br1  = (const float*)d_in[9];
    const float* We1  = (const float*)d_in[10];
    const float* att1 = (const float*)d_in[11];
    const float* bia1 = (const float*)d_in[12];
    const float* Wl2  = (const float*)d_in[13];
    const float* bl2  = (const float*)d_in[14];
    const float* Wr2  = (const float*)d_in[15];
    const float* br2  = (const float*)d_in[16];
    const float* We2  = (const float*)d_in[17];
    const float* att2 = (const float*)d_in[18];
    const float* bia2 = (const float*)d_in[19];
    const float* Wg   = (const float*)d_in[20];
    const float* bg   = (const float*)d_in[21];
    const float* Wfc  = (const float*)d_in[22];
    const float* bfc  = (const float*)d_in[23];
    float* out = (float*)d_out;

    // ---- workspace carve-up ----
    const long NDv = (long)NN * DD;
    float*    xl     = (float*)d_ws;
    float*    xr     = xl + NDv;
    float*    hbuf   = xr + NDv;                       // accumulator, then h
    float*    logits = hbuf + NDv;
    float*    expv   = logits + (long)EE * HH;
    unsigned* segmax = (unsigned*)(expv + (long)EE * HH);
    float*    segsum = (float*)(segmax + (long)NN * HH);
    float*    gl     = segsum + (long)NN * HH;
    float*    ge     = gl + NN;
    unsigned* bmax   = (unsigned*)(ge + NN);
    float*    bsum   = (float*)(bmax + BB);
    float*    pooled = bsum + BB;
    __bf16*   Xb     = (__bf16*)(pooled + (long)BB * DD);   // [N,1024] bf16 (16B aligned)
    __bf16*   Wtb    = Xb + NDv;                            // [Do,K] bf16, transposed weights

    const dim3 blk256(256), blk128(128);
    const dim3 gemmGrid(DD / 128, NN / 16);

#define ZERO(p, n) k_zero<<<(int)(((n) + 255) / 256), blk256, 0, stream>>>((unsigned*)(p), (long)(n))
#define CVT(p, q, n) k_cvt_bf16<<<(int)(((n) + 255) / 256), blk256, 0, stream>>>((p), (q), (long)(n))
#define CVT_T(p, K_) k_cvt_transpose_bf16<<<((K_) * DD + 255) / 256, blk256, 0, stream>>>((p), Wtb, (K_), DD)

    // ============ layer 1 ============
    CVT(x, Xb, (long)NN * XDD);
    CVT_T(Wl1, XDD);
    gemm_bias_wmma<<<gemmGrid, blk128, 0, stream>>>(Xb, Wtb, bl1, xl, NN, XDD, DD);
    CVT_T(Wr1, XDD);
    gemm_bias_wmma<<<gemmGrid, blk128, 0, stream>>>(Xb, Wtb, br1, xr, NN, XDD, DD);
    ZERO(segmax, (long)NN * HH * 2);                   // segmax + segsum (contiguous)
    ZERO(hbuf, NDv);
    k_edge_logits<<<EE, blk256, 0, stream>>>(src, dst, ea, We1, att1, xl, xr, logits, segmax);
    k_edge_exp<<<(EE * HH + 255) / 256, blk256, 0, stream>>>(dst, logits, segmax, expv, segsum);
    k_edge_aggregate<<<EE, blk256, 0, stream>>>(src, dst, expv, segsum, xl, hbuf);
    k_node_finish<<<(int)((NDv + 255) / 256), blk256, 0, stream>>>(hbuf, bia1, NDv);

    // ============ layer 2 ============
    CVT(hbuf, Xb, NDv);
    CVT_T(Wl2, DD);
    gemm_bias_wmma<<<gemmGrid, blk128, 0, stream>>>(Xb, Wtb, bl2, xl, NN, DD, DD);
    CVT_T(Wr2, DD);
    gemm_bias_wmma<<<gemmGrid, blk128, 0, stream>>>(Xb, Wtb, br2, xr, NN, DD, DD);
    ZERO(segmax, (long)NN * HH * 2);
    ZERO(hbuf, NDv);                                   // h no longer needed
    k_edge_logits<<<EE, blk256, 0, stream>>>(src, dst, ea, We2, att2, xl, xr, logits, segmax);
    k_edge_exp<<<(EE * HH + 255) / 256, blk256, 0, stream>>>(dst, logits, segmax, expv, segsum);
    k_edge_aggregate<<<EE, blk256, 0, stream>>>(src, dst, expv, segsum, xl, hbuf);
    k_node_finish<<<(int)((NDv + 255) / 256), blk256, 0, stream>>>(hbuf, bia2, NDv);

    // ============ readout ============
    ZERO(bmax, BB * 2 + (long)BB * DD);                // bmax + bsum + pooled (contiguous)
    k_gate<<<NN / 8, blk256, 0, stream>>>(hbuf, Wg, bg, bat, gl, bmax);
    k_gate_exp<<<(NN + 255) / 256, blk256, 0, stream>>>(bat, gl, bmax, ge, bsum);
    k_pool<<<NN, blk256, 0, stream>>>(bat, ge, bsum, hbuf, pooled);
    k_final<<<BB, blk256, 0, stream>>>(pooled, g, r, Wfc, bfc, out);
#undef ZERO
#undef CVT
#undef CVT_T
}
